// DeepFM_75617194213445
// MI455X (gfx1250) — matrix operand order repaired and checked by reading, not compiled, not run on hardware
//
#include <hip/hip_runtime.h>
#include <stdint.h>

// DeepFM on MI455X (gfx1250): 5 compute-bound GEMMs on the WMMA pipe
// (v_wmma_f32_16x16x32_f16, fp32 accumulate), global->LDS staging via the
// CDNA5 async-DMA path (global_load_async_to_lds_b128 + s_wait_asynccnt)
// to eliminate VGPR staging (which spilled to scratch in the previous rev).

typedef __attribute__((ext_vector_type(16))) _Float16 v16h;
typedef __attribute__((ext_vector_type(8)))  float    v8f;

#define D_DIM 4096
#define B_DIM 8192
#define E_DIM 4096
#define KCHUNK 64
#define LDS_STRIDE 72            // 64 halves + 8 pad -> 144B rows (16B aligned)
#define TILE_HALVES (128 * LDS_STRIDE)

// ---------------------------------------------------------------------------
// CDNA5 async global->LDS copy, 16 bytes per lane, tracked by ASYNCcnt.
// vdst = per-lane LDS byte address (low 32 bits of flat pointer == DS offset),
// vaddr = 64-bit global address.  ISA 15.18.3 opcode 98.
// ---------------------------------------------------------------------------
__device__ __forceinline__ void async_copy_b128(unsigned lds_off, const void* gptr)
{
    asm volatile("global_load_async_to_lds_b128 %0, %1, off"
                 :: "v"(lds_off), "v"(gptr)
                 : "memory");
}
__device__ __forceinline__ void wait_async_le8()
{
    asm volatile("s_wait_asynccnt 0x8" ::: "memory");
}
__device__ __forceinline__ void wait_async_0()
{
    asm volatile("s_wait_asynccnt 0x0" ::: "memory");
}

// ---------------------------------------------------------------------------
// Prep: per-row sum of x, emit x (f16) and relu(x * rowsum) (f16)
// ---------------------------------------------------------------------------
__global__ __launch_bounds__(256)
void prep_x_kernel(const float* __restrict__ x,
                   _Float16* __restrict__ x16,
                   _Float16* __restrict__ inter16)
{
    __shared__ float red[256];
    const int row = blockIdx.x;
    const int tid = threadIdx.x;
    const float* xr = x + (size_t)row * D_DIM;

    float vals[D_DIM / 256];
    float s = 0.f;
#pragma unroll
    for (int i = 0; i < D_DIM / 256; ++i) {
        vals[i] = xr[tid + 256 * i];
        s += vals[i];
    }
    red[tid] = s;
    __syncthreads();
    for (int off = 128; off > 0; off >>= 1) {
        if (tid < off) red[tid] += red[tid + off];
        __syncthreads();
    }
    const float total = red[0];

#pragma unroll
    for (int i = 0; i < D_DIM / 256; ++i) {
        const int c = tid + 256 * i;
        const float v = vals[i];
        x16[(size_t)row * D_DIM + c] = (_Float16)v;
        const float p = v * total;
        inter16[(size_t)row * D_DIM + c] = (_Float16)(p > 0.f ? p : 0.f);
    }
}

// ---------------------------------------------------------------------------
// fp32 -> f16 conversion (weights), 4 elements/thread
// ---------------------------------------------------------------------------
__global__ __launch_bounds__(256)
void cvt_f32_f16_kernel(const float* __restrict__ src,
                        _Float16* __restrict__ dst,
                        long n4)
{
    const long idx = (long)blockIdx.x * blockDim.x + threadIdx.x;
    if (idx >= n4) return;
    const float4 v = ((const float4*)src)[idx];
    _Float16 h[4] = {(_Float16)v.x, (_Float16)v.y, (_Float16)v.z, (_Float16)v.w};
    *(uint2*)(dst + idx * 4) = *(const uint2*)h;
}

// ---------------------------------------------------------------------------
// GEMM: out[b,e] = epilogue( sum_d A[b,d]*W[e,d] + bias[e] )
//   A: [B,D] f16 row-major, W: [E,D] f16 row-major (torch Linear weight)
// MODE 0: relu -> f16
// MODE 1: (relu + inter)*0.5 -> f16   (DNN layer 4 + FM fuse)
// MODE 2: identity -> f32 (final projection into d_out)
// Block tile 128(M)x128(N), 8 waves, wave tile 64x32 (4x2 frags of 16x16).
// Double-buffered LDS filled by async DMA; no VGPR staging.
// ---------------------------------------------------------------------------
template <int MODE>
__global__ __launch_bounds__(256)
void gemm_wmma_kernel(const _Float16* __restrict__ A,
                      const _Float16* __restrict__ W,
                      const float* __restrict__ bias,
                      const _Float16* __restrict__ inter,
                      _Float16* __restrict__ out16,
                      float* __restrict__ out32)
{
    // [buf0 A][buf1 A][buf0 B][buf1 B], each 128*LDS_STRIDE halves
    __shared__ _Float16 lds[4 * TILE_HALVES];

    const int tid  = threadIdx.x;
    const int lane = tid & 31;
    const int wid  = tid >> 5;
    const int blockN = blockIdx.x * 128;
    const int blockM = blockIdx.y * 128;

    const int mWave = (wid >> 2) * 64;  // 2 waves along M
    const int nWave = (wid & 3) * 32;   // 4 waves along N

    v8f acc[4][2];
    const v8f vzero = {0.f, 0.f, 0.f, 0.f, 0.f, 0.f, 0.f, 0.f};
#pragma unroll
    for (int i = 0; i < 4; ++i)
#pragma unroll
        for (int j = 0; j < 2; ++j) acc[i][j] = vzero;

    // per-thread copy slots: 4 x 16B for A tile, 4 x 16B for B tile per chunk
    int rowL[4], kvL[4];
#pragma unroll
    for (int i = 0; i < 4; ++i) {
        const int vecId = tid + 256 * i;
        rowL[i] = vecId >> 3;            // 0..127
        kvL[i]  = (vecId & 7) * 8;       // 0..56 halves
    }

    const unsigned ldsBase = (unsigned)(uintptr_t)&lds[0];

    // issue one chunk's async copies into buffer `buf` (8 insts -> ASYNCcnt += 8)
    auto issue_chunk = [&](int chunk, int buf) {
        const int k0 = chunk * KCHUNK;
#pragma unroll
        for (int i = 0; i < 4; ++i) {
            const unsigned la =
                ldsBase + 2u * ((unsigned)buf * TILE_HALVES +
                                (unsigned)(rowL[i] * LDS_STRIDE + kvL[i]));
            async_copy_b128(la, A + (size_t)(blockM + rowL[i]) * D_DIM + k0 + kvL[i]);
            const unsigned lb =
                ldsBase + 2u * ((unsigned)(2 + buf) * TILE_HALVES +
                                (unsigned)(rowL[i] * LDS_STRIDE + kvL[i]));
            async_copy_b128(lb, W + (size_t)(blockN + rowL[i]) * D_DIM + k0 + kvL[i]);
        }
    };

    const int lhi = lane >> 4;   // half-wave select
    const int llo = lane & 15;

    auto compute_chunk = [&](int buf) {
        const _Float16* Ab = &lds[buf * TILE_HALVES];
        const _Float16* Bb = &lds[(2 + buf) * TILE_HALVES];
#pragma unroll
        for (int ks = 0; ks < KCHUNK; ks += 32) {
            // A frag 16x32 (ISA 7.12.2): lanes 0-15 K {0-7,16-23}, lanes 16-31 K {8-15,24-31}
            v16h afrag[4];
#pragma unroll
            for (int i = 0; i < 4; ++i) {
                const _Float16* base =
                    &Ab[(mWave + 16 * i + llo) * LDS_STRIDE + ks + 8 * lhi];
                *((uint4*)&afrag[i])     = *(const uint4*)(base);
                *((uint4*)&afrag[i] + 1) = *(const uint4*)(base + 16);
            }
            // B frag 32x16: lane = column; lanes 0-15 K {0-7,16-23}, lanes 16-31 K {8-15,24-31}
            v16h bfrag[2];
#pragma unroll
            for (int j = 0; j < 2; ++j) {
                const _Float16* base =
                    &Bb[(nWave + 16 * j + llo) * LDS_STRIDE + ks + 8 * lhi];
                *((uint4*)&bfrag[j])     = *(const uint4*)(base);
                *((uint4*)&bfrag[j] + 1) = *(const uint4*)(base + 16);
            }
#pragma unroll
            for (int i = 0; i < 4; ++i)
#pragma unroll
                for (int j = 0; j < 2; ++j)
                    acc[i][j] = __builtin_amdgcn_wmma_f32_16x16x32_f16(
                        false, afrag[i], false, bfrag[j],
                        (short)0, acc[i][j], false, false);
        }
    };

    constexpr int NCHUNK = D_DIM / KCHUNK;   // 64

    issue_chunk(0, 0);
#pragma unroll 2
    for (int c = 0; c < NCHUNK; ++c) {
        const int cur = c & 1;
        if (c + 1 < NCHUNK) {
            issue_chunk(c + 1, cur ^ 1);   // overlap next DMA with this compute
            wait_async_le8();              // chunk c resident (in-order completion)
        } else {
            wait_async_0();
        }
        __syncthreads();                   // all waves' tiles visible
        compute_chunk(cur);                // 16 x v_wmma per chunk
        if (c + 1 < NCHUNK) __syncthreads(); // all reads done before refill lands
    }

    // Epilogue. C layout: VGPR r, lanes 0-15 -> M=r, lanes 16-31 -> M=r+8; N=lane&15
#pragma unroll
    for (int i = 0; i < 4; ++i) {
#pragma unroll
        for (int j = 0; j < 2; ++j) {
            const int col = blockN + nWave + 16 * j + llo;
            const float bv = bias[col];
#pragma unroll
            for (int r = 0; r < 8; ++r) {
                const int row = blockM + mWave + 16 * i + r + 8 * lhi;
                float v = acc[i][j][r] + bv;
                if (MODE == 0) {
                    v = v > 0.f ? v : 0.f;
                    out16[(size_t)row * E_DIM + col] = (_Float16)v;
                } else if (MODE == 1) {
                    v = v > 0.f ? v : 0.f;
                    const float iv = (float)inter[(size_t)row * E_DIM + col];
                    out16[(size_t)row * E_DIM + col] = (_Float16)((v + iv) * 0.5f);
                } else {
                    out32[(size_t)row * E_DIM + col] = v;
                }
            }
        }
    }
}

// ---------------------------------------------------------------------------
extern "C" void kernel_launch(void* const* d_in, const int* in_sizes, int n_in,
                              void* d_out, int out_size, void* d_ws, size_t ws_size,
                              hipStream_t stream)
{
    const float* x     = (const float*)d_in[0];   // [B, D]
    const float* W_dnn = (const float*)d_in[1];   // [4, D, D]
    const float* b_dnn = (const float*)d_in[2];   // [4, D]
    const float* W_out = (const float*)d_in[3];   // [D, D]
    const float* b_out = (const float*)d_in[4];   // [D]
    float* out = (float*)d_out;                   // [B, D] fp32

    const size_t DD = (size_t)D_DIM * D_DIM;
    const size_t actBytes = (size_t)B_DIM * D_DIM * sizeof(_Float16);

    char* ws = (char*)d_ws;
    _Float16* actA  = (_Float16*)(ws);                 // ping
    _Float16* actB  = (_Float16*)(ws + actBytes);      // pong
    _Float16* inter = (_Float16*)(ws + 2 * actBytes);  // FM term
    _Float16* w16   = (_Float16*)(ws + 3 * actBytes);  // 5*D*D halves

    // 1) prep: x -> f16, interactions -> f16
    prep_x_kernel<<<B_DIM, 256, 0, stream>>>(x, actA, inter);

    // 2) weights -> f16 (once; reused across 64 row-blocks per GEMM)
    {
        const long n4_dnn = (long)(4 * DD / 4);
        cvt_f32_f16_kernel<<<(unsigned)((n4_dnn + 255) / 256), 256, 0, stream>>>(
            W_dnn, w16, n4_dnn);
        const long n4_out = (long)(DD / 4);
        cvt_f32_f16_kernel<<<(unsigned)((n4_out + 255) / 256), 256, 0, stream>>>(
            W_out, w16 + 4 * DD, n4_out);
    }

    // 3) five GEMMs on the WMMA pipe
    dim3 grid(E_DIM / 128, B_DIM / 128);
    gemm_wmma_kernel<0><<<grid, 256, 0, stream>>>(actA, w16 + 0 * DD, b_dnn + 0 * D_DIM,
                                                  nullptr, actB, nullptr);
    gemm_wmma_kernel<0><<<grid, 256, 0, stream>>>(actB, w16 + 1 * DD, b_dnn + 1 * D_DIM,
                                                  nullptr, actA, nullptr);
    gemm_wmma_kernel<0><<<grid, 256, 0, stream>>>(actA, w16 + 2 * DD, b_dnn + 2 * D_DIM,
                                                  nullptr, actB, nullptr);
    gemm_wmma_kernel<1><<<grid, 256, 0, stream>>>(actB, w16 + 3 * DD, b_dnn + 3 * D_DIM,
                                                  inter, actA, nullptr);
    gemm_wmma_kernel<2><<<grid, 256, 0, stream>>>(actA, w16 + 4 * DD, b_out,
                                                  nullptr, nullptr, out);
}
